// Clustered_Attention_Chunking_74294344286859
// MI455X (gfx1250) — matrix operand order repaired and checked by reading, not compiled, not run on hardware
//
#include <hip/hip_runtime.h>
#include <hip/hip_bf16.h>

#define Nn 256
#define Cc 256
#define Ee 256
#define Hh 4
#define DH 64
#define KCL 8
#define CS 32

typedef __attribute__((ext_vector_type(16))) _Float16 v16h;
typedef __attribute__((ext_vector_type(8)))  float    v8f;

__device__ __forceinline__ v8f wmma16(v16h a, v16h b, v8f c) {
    // D = A(16x32 f16) * B(32x16 f16) + C(16x16 f32)
    return __builtin_amdgcn_wmma_f32_16x16x32_f16(false, a, false, b, (short)0, c, false, false);
}

__device__ __forceinline__ int lane_id() { return (int)(threadIdx.x & 31u); }

// ---- fragment loaders (wave32; A and B share contiguous k = 16*hi + e) ----

// A 16x32: lane m=l&15, hi=l>>4; elem e -> K = 16*hi + e
__device__ __forceinline__ v16h afrag_f32(const float* A, int ld, int row0, int k0) {
    int l = lane_id(), m = l & 15, hi = l >> 4;
    const float* p = A + (size_t)(row0 + m) * ld + k0 + hi * 16;
    v16h r;
#pragma unroll
    for (int e = 0; e < 16; ++e) r[e] = (_Float16)p[e];
    return r;
}
__device__ __forceinline__ v16h afrag_f16(const _Float16* A, int ld, int row0, int k0) {
    int l = lane_id(), m = l & 15, hi = l >> 4;
    const _Float16* p = A + (size_t)(row0 + m) * ld + k0 + hi * 16;
    v16h r;
#pragma unroll
    for (int e = 0; e < 16; ++e) r[e] = p[e];
    return r;
}
__device__ __forceinline__ v16h afrag_f16_gather(const _Float16* A, int ld, const int* rows,
                                                 int m0, int k0) {
    int l = lane_id(), m = l & 15, hi = l >> 4;
    const _Float16* p = A + (size_t)rows[m0 + m] * ld + k0 + hi * 16;
    v16h r;
#pragma unroll
    for (int e = 0; e < 16; ++e) r[e] = p[e];
    return r;
}

// B 32x16 with B[k][n] = R[n0+n][k0+k]  (K-contiguous source)
__device__ __forceinline__ v16h bfrag_nmaj_f16(const _Float16* R, int ld, int n0, int k0) {
    int l = lane_id(), n = l & 15, hi = l >> 4;
    const _Float16* p = R + (size_t)(n0 + n) * ld + k0 + hi * 16;
    v16h r;
#pragma unroll
    for (int e = 0; e < 16; ++e) r[e] = p[e];
    return r;
}
__device__ __forceinline__ v16h bfrag_nmaj_f16_gather(const _Float16* R, int ld, const int* rows,
                                                      int n0, int k0) {
    int l = lane_id(), n = l & 15, hi = l >> 4;
    const _Float16* p = R + (size_t)rows[n0 + n] * ld + k0 + hi * 16;
    v16h r;
#pragma unroll
    for (int e = 0; e < 16; ++e) r[e] = p[e];
    return r;
}

// B 32x16 from a TRANSPOSED source RT[n][k] with per-K row gather:
// B[k][n] = RT[(n0+n)*ldT + rows[k0+k]]
__device__ __forceinline__ v16h bfrag_T_gather(const _Float16* RT, int ldT, const int* rows,
                                               int k0, int n0) {
    int l = lane_id(), n = l & 15, hi = l >> 4;
    const _Float16* p = RT + (size_t)(n0 + n) * ldT;
    v16h r;
#pragma unroll
    for (int e = 0; e < 16; ++e) r[e] = p[rows[k0 + hi * 16 + e]];
    return r;
}

// C/D: lane (hi,n) holds rows M = v + 8*hi, col N = n
__device__ __forceinline__ void store_c_f16(_Float16* O, int ld, int row0, int col0, v8f c) {
    int l = lane_id(), n = l & 15, hi = l >> 4;
#pragma unroll
    for (int v = 0; v < 8; ++v)
        O[(size_t)(row0 + v + 8 * hi) * ld + col0 + n] = (_Float16)c[v];
}
// transposed store: O_T[col][row]; contiguous 8 halves per lane
__device__ __forceinline__ void store_c_f16_T(_Float16* OT, int ldT, int row0, int col0, v8f c) {
    int l = lane_id(), n = l & 15, hi = l >> 4;
    _Float16* p = OT + (size_t)(col0 + n) * ldT + row0 + 8 * hi;
#pragma unroll
    for (int v = 0; v < 8; ++v) p[v] = (_Float16)c[v];
}
__device__ __forceinline__ void store_c_f32(float* O, int ld, int row0, int col0, v8f c) {
    int l = lane_id(), n = l & 15, hi = l >> 4;
#pragma unroll
    for (int v = 0; v < 8; ++v)
        O[(size_t)(row0 + v + 8 * hi) * ld + col0 + n] = c[v];
}

// ---------------- kernel 0a: f32 -> f16 bulk convert (8 elements/thread)
__global__ __launch_bounds__(256) void cvt_f16_kernel(const float* __restrict__ in,
                                                      _Float16* __restrict__ out, int n8) {
    int i = blockIdx.x * 256 + (int)threadIdx.x;
    if (i >= n8) return;
    size_t base = (size_t)i * 8;
#pragma unroll
    for (int e = 0; e < 8; ++e) out[base + e] = (_Float16)in[base + e];
}

// ---------------- kernel 0b: stable counting sort per batch (argsort of cluster_id)
__global__ void sort_kernel(const int* __restrict__ cid, int* __restrict__ order) {
    int b = blockIdx.x;
    if (threadIdx.x != 0) return;
    const int* c = cid + (size_t)b * Cc;
    int* o = order + (size_t)b * Cc;
    int cnt[KCL];
#pragma unroll
    for (int k = 0; k < KCL; ++k) cnt[k] = 0;
    for (int i = 0; i < Cc; ++i) cnt[c[i]]++;
    int off[KCL], s = 0;
#pragma unroll
    for (int k = 0; k < KCL; ++k) { off[k] = s; s += cnt[k]; }
    for (int i = 0; i < Cc; ++i) o[off[c[i]]++] = i;
}

// ---------------- kernel 1: fused QKV projection (f16 in, f16 out; V also transposed)
__global__ __launch_bounds__(256) void proj_qkv_kernel(
    const _Float16* __restrict__ seqh,
    const _Float16* __restrict__ Wqh, const float* __restrict__ bq,
    const _Float16* __restrict__ Wkh, const float* __restrict__ bk,
    const _Float16* __restrict__ Wvh, const float* __restrict__ bv,
    _Float16* __restrict__ Qh, _Float16* __restrict__ Kh, _Float16* __restrict__ VhT) {
    int tile = blockIdx.x * 8 + (int)(threadIdx.x >> 5);
    int mt = tile >> 4, nt = tile & 15;
    int m0 = mt * 16, n0 = nt * 16;
    int l = lane_id(), n = l & 15;
    // warm the streaming A tile into cache (global_prefetch_b8)
    __builtin_prefetch(seqh + (size_t)(m0 + (l & 15)) * Ee, 0, 1);
    v8f aq = {}, ak = {}, av = {};
    for (int kt = 0; kt < 8; ++kt) {
        v16h af = afrag_f16(seqh, Ee, m0, kt * 32);
        aq = wmma16(af, bfrag_nmaj_f16(Wqh, Ee, n0, kt * 32), aq);
        ak = wmma16(af, bfrag_nmaj_f16(Wkh, Ee, n0, kt * 32), ak);
        av = wmma16(af, bfrag_nmaj_f16(Wvh, Ee, n0, kt * 32), av);
    }
    float xq = bq[n0 + n], xk = bk[n0 + n], xv = bv[n0 + n];
#pragma unroll
    for (int v = 0; v < 8; ++v) { aq[v] += xq; ak[v] += xk; av[v] += xv; }
    store_c_f16(Qh, Ee, m0, n0, aq);
    store_c_f16(Kh, Ee, m0, n0, ak);
    // V stored transposed per-batch: VhT[b][col][row], row = m0 % Cc within batch b
    int b = m0 / Cc, rloc = m0 - b * Cc;
    store_c_f16_T(VhT + (size_t)b * Ee * Cc, Cc, rloc, n0, av);
}

// ---------------- kernel 2: full attention, one block per (query-tile, head, batch)
__global__ __launch_bounds__(256) void attn_full_kernel(
    const _Float16* __restrict__ Qh, const _Float16* __restrict__ Kh,
    const _Float16* __restrict__ VhT, const float* __restrict__ amask,
    _Float16* __restrict__ ctxF) {
    int qt = blockIdx.x, h = blockIdx.y, b = blockIdx.z;
    int tid = (int)threadIdx.x, wv = tid >> 5, l = tid & 31, n = l & 15, hi = l >> 4;
    int m0 = qt * 16;
    int rowbase = b * Cc;
    const _Float16* VTb = VhT + (size_t)b * Ee * Cc;

    __shared__ float red[8][16];
    __shared__ float Pl[8][16 * 32];
    __shared__ float ctxp[8][16 * 64];

    v16h afr[2];
    afr[0] = afrag_f16(Qh, Ee, rowbase + m0, h * DH);
    afr[1] = afrag_f16(Qh, Ee, rowbase + m0, h * DH + 32);

    v8f sc[2];
#pragma unroll
    for (int t = 0; t < 2; ++t) {
        int nt = wv * 2 + t;
        v8f a = {};
        a = wmma16(afr[0], bfrag_nmaj_f16(Kh, Ee, rowbase + nt * 16, h * DH), a);
        a = wmma16(afr[1], bfrag_nmaj_f16(Kh, Ee, rowbase + nt * 16, h * DH + 32), a);
#pragma unroll
        for (int v = 0; v < 8; ++v) {
            int m = m0 + v + 8 * hi;
            float mk = amask[((size_t)b * Cc + m) * Cc + nt * 16 + n];
            a[v] = a[v] * 0.125f + mk;
        }
        sc[t] = a;
    }

    float gm[8], dn[8];
#pragma unroll
    for (int v = 0; v < 8; ++v) {
        float x = fmaxf(sc[0][v], sc[1][v]);
        x = fmaxf(x, __shfl_xor(x, 1, 32));
        x = fmaxf(x, __shfl_xor(x, 2, 32));
        x = fmaxf(x, __shfl_xor(x, 4, 32));
        x = fmaxf(x, __shfl_xor(x, 8, 32));
        if (n == 0) red[wv][hi * 8 + v] = x;
    }
    __syncthreads();
#pragma unroll
    for (int v = 0; v < 8; ++v) {
        float x = red[0][hi * 8 + v];
#pragma unroll
        for (int w2 = 1; w2 < 8; ++w2) x = fmaxf(x, red[w2][hi * 8 + v]);
        gm[v] = x;
    }
    __syncthreads();
#pragma unroll
    for (int v = 0; v < 8; ++v) {
        sc[0][v] = __expf(sc[0][v] - gm[v]);
        sc[1][v] = __expf(sc[1][v] - gm[v]);
        float s = sc[0][v] + sc[1][v];
        s += __shfl_xor(s, 1, 32);
        s += __shfl_xor(s, 2, 32);
        s += __shfl_xor(s, 4, 32);
        s += __shfl_xor(s, 8, 32);
        if (n == 0) red[wv][hi * 8 + v] = s;
    }
    __syncthreads();
#pragma unroll
    for (int v = 0; v < 8; ++v) {
        float s = 0.f;
#pragma unroll
        for (int w2 = 0; w2 < 8; ++w2) s += red[w2][hi * 8 + v];
        dn[v] = 1.0f / s;
    }
    // stage probabilities for this wave's 32-key slice, then P x V
#pragma unroll
    for (int t = 0; t < 2; ++t)
#pragma unroll
        for (int v = 0; v < 8; ++v)
            Pl[wv][(v + 8 * hi) * 32 + t * 16 + n] = sc[t][v] * dn[v];
    __syncthreads();

    v16h pa = afrag_f32(&Pl[wv][0], 32, 0, 0);
#pragma unroll
    for (int nt2 = 0; nt2 < 4; ++nt2) {
        v8f z = {};
        // B[k][n] = V[k][col] = VTb[(h*DH+nt2*16+n)*Cc + k], contiguous along k
        v8f cx = wmma16(pa, bfrag_nmaj_f16(VTb, Cc, h * DH + nt2 * 16, wv * 32), z);
#pragma unroll
        for (int v = 0; v < 8; ++v)
            ctxp[wv][(v + 8 * hi) * 64 + nt2 * 16 + n] = cx[v];
    }
    __syncthreads();
    for (int e = tid; e < 16 * 64; e += 256) {
        float s = 0.f;
#pragma unroll
        for (int w2 = 0; w2 < 8; ++w2) s += ctxp[w2][e];
        int m = e >> 6, c = e & 63;
        ctxF[((size_t)b * Cc + m0 + m) * Ee + h * DH + c] = (_Float16)s;
    }
}

// ---------------- kernel 3: chunked attention, one block per (chunk, batch); wave = (head, mtile)
__global__ __launch_bounds__(256) void attn_chunk_kernel(
    const _Float16* __restrict__ Qh, const _Float16* __restrict__ Kh,
    const _Float16* __restrict__ VhT, const float* __restrict__ amask,
    const int* __restrict__ order, _Float16* __restrict__ ctxC) {
    int ch = blockIdx.x, b = blockIdx.y;
    int tid = (int)threadIdx.x, wv = tid >> 5, l = tid & 31, n = l & 15, hi = l >> 4;
    int h = wv >> 1, mt = wv & 1;

    __shared__ int kidx[2 * CS];
    __shared__ int qidx[CS];
    __shared__ float Pl[8][16 * 64];

    const int* ord = order + (size_t)b * Cc;
    int ks0 = (ch < 2) ? 0 : (ch - 1) * CS;
    if (tid < 2 * CS) kidx[tid] = ord[ks0 + tid];
    if (tid < CS) qidx[tid] = ord[ch * CS + tid];
    __syncthreads();

    const _Float16* Qb = Qh + (size_t)b * Cc * Ee;
    const _Float16* Kb = Kh + (size_t)b * Cc * Ee;
    const _Float16* VTb = VhT + (size_t)b * Ee * Cc;

    v16h aq[2];
    aq[0] = afrag_f16_gather(Qb, Ee, qidx, mt * 16, h * DH);
    aq[1] = afrag_f16_gather(Qb, Ee, qidx, mt * 16, h * DH + 32);

    v8f sc[4];
#pragma unroll
    for (int nt = 0; nt < 4; ++nt) {
        v8f a = {};
        a = wmma16(aq[0], bfrag_nmaj_f16_gather(Kb, Ee, kidx, nt * 16, h * DH), a);
        a = wmma16(aq[1], bfrag_nmaj_f16_gather(Kb, Ee, kidx, nt * 16, h * DH + 32), a);
#pragma unroll
        for (int v = 0; v < 8; ++v) {
            float mk = 0.f;
            if (nt < 2)
                mk = amask[((size_t)b * Cc + ch * CS + mt * 16 + v + 8 * hi) * Cc +
                           ch * CS + nt * 16 + n];
            a[v] = a[v] * 0.125f + mk;
        }
        sc[nt] = a;
    }
    // in-wave softmax over 64 keys
#pragma unroll
    for (int v = 0; v < 8; ++v) {
        float x = fmaxf(fmaxf(sc[0][v], sc[1][v]), fmaxf(sc[2][v], sc[3][v]));
        x = fmaxf(x, __shfl_xor(x, 1, 32));
        x = fmaxf(x, __shfl_xor(x, 2, 32));
        x = fmaxf(x, __shfl_xor(x, 4, 32));
        x = fmaxf(x, __shfl_xor(x, 8, 32));
        float s = 0.f;
#pragma unroll
        for (int nt = 0; nt < 4; ++nt) { sc[nt][v] = __expf(sc[nt][v] - x); s += sc[nt][v]; }
        s += __shfl_xor(s, 1, 32);
        s += __shfl_xor(s, 2, 32);
        s += __shfl_xor(s, 4, 32);
        s += __shfl_xor(s, 8, 32);
        float rp = 1.0f / s;
#pragma unroll
        for (int nt = 0; nt < 4; ++nt)
            Pl[wv][(v + 8 * hi) * 64 + nt * 16 + n] = sc[nt][v] * rp;
    }
    __syncthreads();

    v16h pa0 = afrag_f32(&Pl[wv][0], 64, 0, 0);
    v16h pa1 = afrag_f32(&Pl[wv][0], 64, 0, 32);
#pragma unroll
    for (int nt2 = 0; nt2 < 4; ++nt2) {
        v8f a = {};
        a = wmma16(pa0, bfrag_T_gather(VTb, Cc, kidx, 0, h * DH + nt2 * 16), a);
        a = wmma16(pa1, bfrag_T_gather(VTb, Cc, kidx, 32, h * DH + nt2 * 16), a);
        store_c_f16(ctxC + (size_t)b * Cc * Ee, Ee, ch * CS + mt * 16, h * DH + nt2 * 16, a);
    }
}

// ---------------- kernel 4: output projection  hid = ctx @ Wd^T + bd (f32 result)
__global__ __launch_bounds__(256) void proj_d_kernel(
    const _Float16* __restrict__ X, const _Float16* __restrict__ Wdh,
    const float* __restrict__ bd, float* __restrict__ hid) {
    int tile = blockIdx.x * 8 + (int)(threadIdx.x >> 5);
    int mt = tile >> 4, nt = tile & 15;
    int m0 = mt * 16, n0 = nt * 16;
    int l = lane_id(), n = l & 15;
    __builtin_prefetch(X + (size_t)(m0 + (l & 15)) * Ee, 0, 1);
    v8f a = {};
    for (int kt = 0; kt < 8; ++kt)
        a = wmma16(afrag_f16(X, Ee, m0, kt * 32), bfrag_nmaj_f16(Wdh, Ee, n0, kt * 32), a);
    float bb = bd[n0 + n];
#pragma unroll
    for (int v = 0; v < 8; ++v) a[v] += bb;
    store_c_f32(hid, Ee, m0, n0, a);
}

// ---------------- kernel 5: residual + layernorm both paths, unsort, blend
__global__ __launch_bounds__(256) void finalize_kernel(
    const float* __restrict__ seq, const float* __restrict__ hidF,
    const float* __restrict__ hidC, const int* __restrict__ order,
    const float* __restrict__ lnw, const float* __restrict__ lnb,
    float* __restrict__ out) {
    int b = blockIdx.y;
    int r = blockIdx.x * 8 + (int)(threadIdx.x >> 5);  // sorted row
    int l = lane_id();
    int i = order[(size_t)b * Cc + r];                 // original row
    const float* sq = seq + ((size_t)b * Cc + i) * Ee;
    const float* hf = hidF + ((size_t)b * Cc + i) * Ee;
    const float* hc = hidC + ((size_t)b * Cc + r) * Ee;
    float xs[8], ys[8];
    float sx = 0.f, sxx = 0.f, sy = 0.f, syy = 0.f;
#pragma unroll
    for (int e = 0; e < 8; ++e) {
        int c = l * 8 + e;
        float x = hf[c] + sq[c];
        float y = hc[c] + sq[c];
        xs[e] = x; ys[e] = y;
        sx += x; sxx += x * x; sy += y; syy += y * y;
    }
#pragma unroll
    for (int m = 1; m < 32; m <<= 1) {
        sx += __shfl_xor(sx, m, 32);
        sxx += __shfl_xor(sxx, m, 32);
        sy += __shfl_xor(sy, m, 32);
        syy += __shfl_xor(syy, m, 32);
    }
    const float inv = 1.0f / (float)Ee;
    float ux = sx * inv, vx = sxx * inv - ux * ux;
    float uy = sy * inv, vy = syy * inv - uy * uy;
    float rx = rsqrtf(vx + 1e-12f);
    float ry = rsqrtf(vy + 1e-12f);
    float* o = out + ((size_t)b * Cc + i) * Ee;
#pragma unroll
    for (int e = 0; e < 8; ++e) {
        int c = l * 8 + e;
        float w = lnw[c], bb = lnb[c];
        float fo = w * (xs[e] - ux) * rx + bb;
        float co = w * (ys[e] - uy) * ry + bb;
        o[c] = 0.5f * fo + 0.5f * co;
    }
}

extern "C" void kernel_launch(void* const* d_in, const int* in_sizes, int n_in,
                              void* d_out, int out_size, void* d_ws, size_t ws_size,
                              hipStream_t stream) {
    const float* seq   = (const float*)d_in[0];
    const float* amask = (const float*)d_in[1];
    const int*   cid   = (const int*)d_in[2];
    const float* Wq = (const float*)d_in[3];  const float* bq = (const float*)d_in[4];
    const float* Wk = (const float*)d_in[5];  const float* bk = (const float*)d_in[6];
    const float* Wv = (const float*)d_in[7];  const float* bv = (const float*)d_in[8];
    const float* Wd = (const float*)d_in[9];  const float* bd = (const float*)d_in[10];
    const float* lnw = (const float*)d_in[11];
    const float* lnb = (const float*)d_in[12];
    float* out = (float*)d_out;

    const size_t NT = (size_t)Nn * Cc;
    const size_t EW = (size_t)Ee * Ee;
    char* p = (char*)d_ws;
    auto take = [&](size_t bytes) { char* q = p; p += (bytes + 255) & ~(size_t)255; return q; };
    int*       order = (int*)take(sizeof(int) * NT);
    _Float16*  seqh  = (_Float16*)take(2ull * NT * Ee);
    _Float16*  Wqh   = (_Float16*)take(2ull * EW);
    _Float16*  Wkh   = (_Float16*)take(2ull * EW);
    _Float16*  Wvh   = (_Float16*)take(2ull * EW);
    _Float16*  Wdh   = (_Float16*)take(2ull * EW);
    _Float16*  Qh    = (_Float16*)take(2ull * NT * Ee);
    _Float16*  Kh    = (_Float16*)take(2ull * NT * Ee);
    _Float16*  VhT   = (_Float16*)take(2ull * NT * Ee);
    _Float16*  ctxF  = (_Float16*)take(2ull * NT * Ee);
    _Float16*  ctxC  = (_Float16*)take(2ull * NT * Ee);
    float*     hidF  = (float*)take(4ull * NT * Ee);
    float*     hidC  = (float*)take(4ull * NT * Ee);
    (void)ws_size; (void)in_sizes; (void)n_in; (void)out_size;

    // precision staging: f32 -> f16 once (removes per-tile cvt redundancy)
    int n8_seq = (int)(NT * Ee / 8);
    int n8_w   = (int)(EW / 8);
    cvt_f16_kernel<<<(n8_seq + 255) / 256, 256, 0, stream>>>(seq, seqh, n8_seq);
    cvt_f16_kernel<<<(n8_w + 255) / 256, 256, 0, stream>>>(Wq, Wqh, n8_w);
    cvt_f16_kernel<<<(n8_w + 255) / 256, 256, 0, stream>>>(Wk, Wkh, n8_w);
    cvt_f16_kernel<<<(n8_w + 255) / 256, 256, 0, stream>>>(Wv, Wvh, n8_w);
    cvt_f16_kernel<<<(n8_w + 255) / 256, 256, 0, stream>>>(Wd, Wdh, n8_w);

    sort_kernel<<<Nn, 32, 0, stream>>>(cid, order);

    int gemm_blocks = (int)((NT / 16) * (Ee / 16) / 8);  // 8 waves/block, 1 tile/wave
    proj_qkv_kernel<<<gemm_blocks, 256, 0, stream>>>(seqh, Wqh, bq, Wkh, bk, Wvh, bv,
                                                     Qh, Kh, VhT);

    dim3 gf(Cc / 16, Hh, Nn);
    attn_full_kernel<<<gf, 256, 0, stream>>>(Qh, Kh, VhT, amask, ctxF);

    dim3 gc(KCL, Nn);
    attn_chunk_kernel<<<gc, 256, 0, stream>>>(Qh, Kh, VhT, amask, order, ctxC);

    proj_d_kernel<<<gemm_blocks, 256, 0, stream>>>(ctxF, Wdh, bd, hidF);
    proj_d_kernel<<<gemm_blocks, 256, 0, stream>>>(ctxC, Wdh, bd, hidC);

    dim3 g5(Cc / 8, Nn);
    finalize_kernel<<<g5, 256, 0, stream>>>(seq, hidF, hidC, order, lnw, lnb, out);
}
